// MultiHeadSelfAttention_72748156059791
// MI455X (gfx1250) — matrix operand order repaired and checked by reading, not compiled, not run on hardware
//
#include <hip/hip_runtime.h>

// ---------------- types & helpers ----------------
typedef __attribute__((ext_vector_type(16))) __bf16 v16bf;
typedef __attribute__((ext_vector_type(8)))  float  v8f;

#define NEG_INF (-__builtin_inff())

__device__ __forceinline__ unsigned short f32_to_bf16(float f) {
    union { float f; unsigned int u; } x; x.f = f;
    unsigned int u = x.u;
    unsigned int r = (u + 0x7FFFu + ((u >> 16) & 1u)) >> 16;  // RNE
    return (unsigned short)r;
}
__device__ __forceinline__ float bf16_to_f32(unsigned short h) {
    union { unsigned int u; float f; } x; x.u = ((unsigned int)h) << 16;
    return x.f;
}

union FragU { v16bf v; uint4 q[2]; };

// A fragment (16x32 bf16, row-major source, lda in elements)
// lane: M = lane%16, half = lane/16; VGPR0-3 -> K = half*8+0..7, VGPR4-7 -> K = 16+half*8+0..7
__device__ __forceinline__ v16bf load_a_frag(const unsigned short* A, int lda,
                                             int row0, int k0, int lane) {
    int m = row0 + (lane & 15);
    int half = lane >> 4;
    const unsigned short* p = A + (size_t)m * lda + k0 + half * 8;
    FragU u;
    u.q[0] = *(const uint4*)(p);
    u.q[1] = *(const uint4*)(p + 16);
    return u.v;
}

// B fragment (32x16 bf16, row-major source [K x N], ldb in elements)
// lane: N = lane%16, half = lane/16; elems 0-7 -> K = half*8+0..7, elems 8-15 -> K = 16+half*8+0..7
__device__ __forceinline__ v16bf load_b_frag(const unsigned short* B, int ldb,
                                             int k0, int col0, int lane) {
    int n = col0 + (lane & 15);
    int half = lane >> 4;
    FragU u;
    for (int s = 0; s < 16; ++s) {
        int k = (s < 8) ? (half * 8 + s) : (16 + half * 8 + (s - 8));
        u.v[s] = __builtin_bit_cast(__bf16, B[(size_t)(k0 + k) * ldb + n]);
    }
    return u.v;
}

__device__ __forceinline__ v8f wmma_bf16(v16bf a, v16bf b, v8f c) {
    return __builtin_amdgcn_wmma_f32_16x16x32_bf16(false, a, false, b,
                                                   (short)0, c, false, false);
}

// ---- CDNA5 async global->LDS copy (ASYNCcnt-tracked), ISA 08_async_tensor §4 ----
template <int OFF>
__device__ __forceinline__ void async_b128(unsigned lds_off, const void* gaddr) {
    asm volatile("global_load_async_to_lds_b128 %0, %1, off offset:%2"
                 :: "v"(lds_off), "v"(gaddr), "i"(OFF) : "memory");
}
#define WAIT_ASYNC(n) asm volatile("s_wait_asynccnt %0" :: "i"(n) : "memory")

// ---------------- problem sizes ----------------
#define BB 4
#define SS 1024
#define EE 1024
#define NH 16
#define DD 64
#define BS (BB*SS)          // 4096

// LDS slab: 32 K-rows x 64 N-cols bf16, rows padded to 72 elems (144B)
#define LDS_ROW   72
#define SLAB_EL   (32 * LDS_ROW)   // elements per buffer
#define SLAB_BY   (SLAB_EL * 2)    // bytes per buffer (4608)

// Shared GEMM body: C[m0..m0+15, col0..col0+63] += A[:,0:1024] * Bsrc[0:1024, ...]
// 4 waves/block cooperate: B slabs staged through LDS with double-buffered
// async copies; A fragments loaded directly (rows are wave-private).
__device__ __forceinline__ void gemm_k1024(const unsigned short* __restrict__ A,
                                           const unsigned short* __restrict__ Bsrc,
                                           int ldb, int col0,
                                           unsigned short* Bs,  // __shared__ [2][32][72]
                                           int m0, v8f acc[4]) {
    const int tid  = threadIdx.x;
    const int lane = tid & 31;
    const int lr = tid >> 2;    // slab row this thread copies (0..31)
    const int lc = tid & 3;     // 32B chunk within row (0..3)
    const unsigned ldsbase = (unsigned)(size_t)Bs + (unsigned)(lr * (LDS_ROW * 2) + lc * 32);
    const unsigned short* gbase = Bsrc + (size_t)lr * ldb + col0 + lc * 16;

    // prefetch slab 0 into buffer 0 (2 async ops per thread -> 2 per wave)
    async_b128<0>(ldsbase, gbase);
    async_b128<16>(ldsbase, gbase);

    // no-unroll: keep a single loop body so the WMMA accumulators are updated
    // in place (dst==src) instead of ping-ponging across register banks
#pragma clang loop unroll(disable)
    for (int it = 0; it < 32; ++it) {
        if (it + 1 < 32) {
            const unsigned short* g = gbase + (size_t)(it + 1) * 32 * ldb;
            unsigned l = ldsbase + ((it + 1) & 1) * SLAB_BY;
            async_b128<0>(l, g);
            async_b128<16>(l, g);
            WAIT_ASYNC(2);          // current slab done; next slab in flight
        } else {
            WAIT_ASYNC(0);
        }
        __syncthreads();            // all waves' slab data visible

        v16bf a = load_a_frag(A, 1024, m0, it * 32, lane);
        const unsigned short* brow = Bs + (it & 1) * SLAB_EL + (lane & 15);
        // hoist all four B fragments so the ds_loads issue as one clause and
        // the dscnt waits overlap with WMMA execution
        FragU u[4];
#pragma unroll
        for (int j = 0; j < 4; ++j) {
            for (int s = 0; s < 16; ++s) {
                int k = (s < 8) ? ((lane >> 4) * 8 + s) : (16 + (lane >> 4) * 8 + (s - 8));
                u[j].v[s] = __builtin_bit_cast(__bf16, brow[k * LDS_ROW + j * 16]);
            }
        }
#pragma unroll
        for (int j = 0; j < 4; ++j)
            acc[j] = wmma_bf16(a, u[j].v, acc[j]);
        __syncthreads();            // protect buffer before next overwrite
    }
}

// ---------------- stage 0: f32 -> bf16 convert ----------------
__global__ void cvt_kernel(const float* __restrict__ in,
                           unsigned short* __restrict__ out, int n) {
    int i = blockIdx.x * blockDim.x + threadIdx.x;
    if (i < n) out[i] = f32_to_bf16(in[i]);
}

// ---------------- stage 1: projections (Q^T, K, Dbuf) ----------------
// grid: (64 m-blocks, 48 combos = head*3types), block 128 (4 waves, 16 rows each)
__global__ __launch_bounds__(128)
void proj_kernel(const unsigned short* __restrict__ Xbf,
                 const unsigned short* __restrict__ Wq,
                 const unsigned short* __restrict__ Wk,
                 const unsigned short* __restrict__ Wvd,
                 unsigned short* __restrict__ QT,    // [NH][DD][BS]
                 unsigned short* __restrict__ Kout,  // [NH][BS][DD]
                 unsigned short* __restrict__ Dbuf)  // [BS][NH*DD]
{
    __shared__ unsigned short Bs[2][32][LDS_ROW];
    int lane = threadIdx.x & 31, wave = threadIdx.x >> 5;
    int combo = blockIdx.y;
    int n = combo & 15, t = combo >> 4;
    const unsigned short* W =
        (t == 0 ? Wq : (t == 1 ? Wk : Wvd)) + (size_t)n * EE * DD;
    int m0 = blockIdx.x * 64 + wave * 16;

    v8f acc[4] = {};
    gemm_k1024(Xbf, W, DD, 0, &Bs[0][0][0], m0, acc);

    int half = lane >> 4, nl = lane & 15;
    if (t == 0) {
        // transposed store: QT_n[d][bs], consecutive in r -> 16B packed store
        unsigned short* Q = QT + (size_t)n * DD * BS;
        for (int j = 0; j < 4; ++j) {
            int d = j * 16 + nl;
            union { unsigned short s[8]; uint4 q; } tmp;
            for (int r = 0; r < 8; ++r) tmp.s[r] = f32_to_bf16(acc[j][r]);
            *(uint4*)(Q + (size_t)d * BS + m0 + 8 * half) = tmp.q;
        }
    } else if (t == 1) {
        unsigned short* K = Kout + (size_t)n * BS * DD;
        for (int j = 0; j < 4; ++j)
            for (int r = 0; r < 8; ++r)
                K[(size_t)(m0 + r + 8 * half) * DD + j * 16 + nl] =
                    f32_to_bf16(acc[j][r]);
    } else {
        for (int j = 0; j < 4; ++j)
            for (int r = 0; r < 8; ++r)
                Dbuf[(size_t)(m0 + r + 8 * half) * (NH * DD) + n * DD + j * 16 + nl] =
                    f32_to_bf16(acc[j][r]);
    }
}

// ---------------- stage 2: causal diag-softmax via WMMA + online stats ----------
// grid: (16, B, NH), block 128; wave handles one 16-row tile
__global__ __launch_bounds__(128)
void diag_softmax_kernel(const unsigned short* __restrict__ Kbf,
                         const unsigned short* __restrict__ QTbf,
                         float* __restrict__ diag)   // [NH][BS]
{
    int lane = threadIdx.x & 31, wave = threadIdx.x >> 5;
    int rt = blockIdx.x * 4 + wave;   // row tile 0..63
    int b  = blockIdx.y;
    int n  = blockIdx.z;
    const unsigned short* Kn = Kbf  + (size_t)n * BS * DD;
    const unsigned short* Qn = QTbf + (size_t)n * DD * BS;
    int m0 = rt * 16;
    int half = lane >> 4, nl = lane & 15;

    v16bf a0 = load_a_frag(Kn, DD, b * SS + m0, 0, lane);
    v16bf a1 = load_a_frag(Kn, DD, b * SS + m0, 32, lane);

    float rmax[8], rsum[8];
    for (int r = 0; r < 8; ++r) { rmax[r] = NEG_INF; rsum[r] = 0.f; }
    float dval = NEG_INF;
    int rr = nl - 8 * half;            // diag-owner row index if in [0,8)

    for (int ct = rt; ct < 64; ++ct) {
        int c0 = ct * 16;
        v16bf b0 = load_b_frag(Qn, BS, 0, b * SS + c0, lane);
        v16bf b1 = load_b_frag(Qn, BS, 32, b * SS + c0, lane);
        v8f acc = {};
        acc = wmma_bf16(a0, b0, acc);
        acc = wmma_bf16(a1, b1, acc);
        int Sg = c0 + nl;              // this lane's column
        for (int r = 0; r < 8; ++r) {
            float val = acc[r] * 0.125f;                  // /sqrt(64)
            int sg = m0 + r + 8 * half;                   // row
            bool masked = (Sg < sg) || (val == 0.0f);     // triu + ==0 quirk
            float v = masked ? NEG_INF : val;
            if (ct == rt && rr == r) dval = v;
            float nm = fmaxf(rmax[r], v);
            if (nm != NEG_INF) {
                float p = (v == NEG_INF) ? 0.0f : __expf(v - nm);
                rsum[r] = rsum[r] * __expf(rmax[r] - nm) + p;
                rmax[r] = nm;
            }
        }
    }
    // merge (max,sum) across the 16 lanes of each half
    for (int r = 0; r < 8; ++r) {
        float mx = rmax[r], sm = rsum[r];
        for (int mask = 1; mask < 16; mask <<= 1) {
            float omx = __shfl_xor(mx, mask, 32);
            float osm = __shfl_xor(sm, mask, 32);
            float nm = fmaxf(mx, omx);
            float s = 0.f;
            if (mx  != NEG_INF) s += sm  * __expf(mx  - nm);
            if (omx != NEG_INF) s += osm * __expf(omx - nm);
            mx = nm; sm = s;
        }
        rmax[r] = mx; rsum[r] = sm;
    }
    if (rr >= 0 && rr < 8) {
        float attn = 0.f;
        if (dval != NEG_INF && rsum[rr] > 0.f)
            attn = __expf(dval - rmax[rr]) / rsum[rr];
        diag[(size_t)n * BS + b * SS + m0 + rr + 8 * half] = attn;
    }
}

// ---------------- stage 2.5: scale Dbuf rows by per-head diag ----------------
__global__ void scale_down_kernel(unsigned short* __restrict__ Dbuf,
                                  const float* __restrict__ diag) {
    int idx = blockIdx.x * blockDim.x + threadIdx.x;   // 0..BS*NH*DD-1
    int m = idx >> 10;             // row (global bs)
    int c = idx & 1023;            // col = n*64+d
    int n = c >> 6;
    float v = bf16_to_f32(Dbuf[idx]) * diag[(size_t)n * BS + m];
    Dbuf[idx] = f32_to_bf16(v);
}

// ---------------- stage 3: delta GEMM + head-mean + residual ----------------
// grid: (64, 16), block 128; wave computes 16x64 tile, K = 1024 (all heads)
__global__ __launch_bounds__(128)
void out_kernel(const unsigned short* __restrict__ Dbuf,   // [BS][1024]
                const unsigned short* __restrict__ WVu,    // [1024][EE]
                const float* __restrict__ inp,
                float* __restrict__ out) {
    __shared__ unsigned short Bs[2][32][LDS_ROW];
    int lane = threadIdx.x & 31, wave = threadIdx.x >> 5;
    int m0 = blockIdx.x * 64 + wave * 16;
    int e0 = blockIdx.y * 64;

    v8f acc[4] = {};
    gemm_k1024(Dbuf, WVu, EE, e0, &Bs[0][0][0], m0, acc);

    int half = lane >> 4, nl = lane & 15;
    for (int j = 0; j < 4; ++j)
        for (int r = 0; r < 8; ++r) {
            size_t idx = (size_t)(m0 + r + 8 * half) * EE + e0 + j * 16 + nl;
            out[idx] = inp[idx] + acc[j][r] * (1.0f / 16.0f);  // mean over heads
        }
}

// ---------------- launch ----------------
extern "C" void kernel_launch(void* const* d_in, const int* in_sizes, int n_in,
                              void* d_out, int out_size, void* d_ws, size_t ws_size,
                              hipStream_t stream) {
    const float* inp  = (const float*)d_in[0];   // [B,S,E]
    const float* WQ   = (const float*)d_in[1];   // [N,E,D]
    const float* WK   = (const float*)d_in[2];   // [N,E,D]
    const float* WVd  = (const float*)d_in[3];   // [N,E,D]
    const float* WVu  = (const float*)d_in[4];   // [N,D,E]
    float* out = (float*)d_out;

    char* ws = (char*)d_ws;
    size_t off = 0;
    unsigned short* Xbf   = (unsigned short*)(ws + off); off += (size_t)BS * EE * 2;        // 8 MB
    unsigned short* WQbf  = (unsigned short*)(ws + off); off += (size_t)NH * EE * DD * 2;   // 2 MB
    unsigned short* WKbf  = (unsigned short*)(ws + off); off += (size_t)NH * EE * DD * 2;
    unsigned short* WVdbf = (unsigned short*)(ws + off); off += (size_t)NH * EE * DD * 2;
    unsigned short* WVubf = (unsigned short*)(ws + off); off += (size_t)NH * DD * EE * 2;
    unsigned short* QTbf  = (unsigned short*)(ws + off); off += (size_t)NH * DD * BS * 2;   // 8 MB
    unsigned short* Kbf   = (unsigned short*)(ws + off); off += (size_t)NH * BS * DD * 2;   // 8 MB
    unsigned short* Dbuf  = (unsigned short*)(ws + off); off += (size_t)BS * NH * DD * 2;   // 8 MB
    float*          diag  = (float*)(ws + off);          off += (size_t)NH * BS * 4;        // 256 KB

    const int inN = BS * EE;            // 4,194,304
    const int wN  = NH * EE * DD;       // 1,048,576
    cvt_kernel<<<(inN + 255) / 256, 256, 0, stream>>>(inp, Xbf, inN);
    cvt_kernel<<<(wN  + 255) / 256, 256, 0, stream>>>(WQ,  WQbf,  wN);
    cvt_kernel<<<(wN  + 255) / 256, 256, 0, stream>>>(WK,  WKbf,  wN);
    cvt_kernel<<<(wN  + 255) / 256, 256, 0, stream>>>(WVd, WVdbf, wN);
    cvt_kernel<<<(wN  + 255) / 256, 256, 0, stream>>>(WVu, WVubf, wN);

    proj_kernel<<<dim3(BS / 64, 48), 128, 0, stream>>>(Xbf, WQbf, WKbf, WVdbf,
                                                       QTbf, Kbf, Dbuf);
    diag_softmax_kernel<<<dim3(16, BB, NH), 128, 0, stream>>>(Kbf, QTbf, diag);
    scale_down_kernel<<<(BS * NH * DD) / 256, 256, 0, stream>>>(Dbuf, diag);
    out_kernel<<<dim3(BS / 64, EE / 64), 128, 0, stream>>>(Dbuf, WVubf, inp, out);
}
